// Decoder_18365280158001
// MI455X (gfx1250) — compile-verified, matrix operand order checked by reading, not run on hardware
//
#include <hip/hip_runtime.h>

#define NODES 8192
#define ZFEAT 128
#define XFEAT 128
#define HDIM  256
#define NEDGE 262144

typedef __attribute__((ext_vector_type(16))) _Float16 v16h;
typedef __attribute__((ext_vector_type(8)))  _Float16 v8h;
typedef __attribute__((ext_vector_type(8)))  float    v8f;
typedef __attribute__((ext_vector_type(4)))  float    v4f;
typedef __attribute__((ext_vector_type(4)))  int      v4i;

// --------------------------------------------------------------------------
// new_edge_index fill: rows = i>>13, cols = i&8191 (N = 8192 = 2^13).
// 512 MB write-once stream -> nontemporal stores keep it out of L2 so the
// aggregation working set (H/AGG, ~16 MB) stays L2-resident.
// --------------------------------------------------------------------------
__global__ void fill_edge_index(int* __restrict__ ei) {
    const int total = NODES * NODES;          // 1<<26, fits int
    const int nvec  = total >> 2;             // v4i elements per half
    v4i* __restrict__ rows = (v4i*)ei;
    v4i* __restrict__ cols = (v4i*)(ei + total);
    int stride = gridDim.x * blockDim.x;
    for (int v = blockIdx.x * blockDim.x + threadIdx.x; v < nvec; v += stride) {
        int base = v << 2;
        int rv = base >> 13;                  // same row across the quad (4 | 8192)
        v4i r = {rv, rv, rv, rv};
        __builtin_nontemporal_store(r, &rows[v]);
        int c0 = base & 8191;                 // no wrap inside the quad
        v4i c = {c0, c0 + 1, c0 + 2, c0 + 3};
        __builtin_nontemporal_store(c, &cols[v]);
    }
}

// --------------------------------------------------------------------------
// Degree: deg[i] = 1 (self loop) + sum_{e: dst==i} w[e];  then dinv = rsqrt.
// --------------------------------------------------------------------------
__global__ void deg_init(float* __restrict__ deg) {
    int i = blockIdx.x * blockDim.x + threadIdx.x;
    if (i < NODES) deg[i] = 1.0f;
}
__global__ void deg_accum(const int* __restrict__ dst, const float* __restrict__ w,
                          float* __restrict__ deg) {
    int e = blockIdx.x * blockDim.x + threadIdx.x;
    if (e < NEDGE) atomicAdd(&deg[dst[e]], w[e]);
}
__global__ void deg_finalize(float* __restrict__ deg) {
    int i = blockIdx.x * blockDim.x + threadIdx.x;
    if (i < NODES) {
        float d = deg[i];
        deg[i] = (d > 0.0f) ? rsqrtf(d) : 0.0f;   // now holds dinv
    }
}

// --------------------------------------------------------------------------
// One-shot conversions feeding the WMMA GEMMs:
//   cvt_f16:        X f32 [n]       -> Xh f16 [n]        (row-major A operand)
//   cvt_wt_f16T:    W  f32 [K,N]    -> Wt f16 [N,K]      (col-major B operand)
// --------------------------------------------------------------------------
__global__ void cvt_f16(const float* __restrict__ x, _Float16* __restrict__ xh, int n) {
    int i = blockIdx.x * blockDim.x + threadIdx.x;
    if (i < n) xh[i] = (_Float16)x[i];
}
__global__ void cvt_wt_f16T(const float* __restrict__ w, _Float16* __restrict__ wt,
                            int K, int N) {
    int i = blockIdx.x * blockDim.x + threadIdx.x;   // over K*N
    if (i < K * N) {
        int k = i / N, n = i - k * N;
        wt[(size_t)n * K + k] = (_Float16)w[i];
    }
}

// --------------------------------------------------------------------------
// WMMA GEMM: Y[M,N] = Xh[M,K] @ Wt[N,K]^T (+ bias).  One wave per 16x16 tile.
// K is a template param (128 or 256) so the K-loop fully unrolls into
// back-to-back v_wmma_f32_16x16x32_f16 with all loads hoisted/pipelined.
// Fragment layout per ISA 7.12.2: lane L (m = L&15, hi = L>>4) holds
// elements e with K(e) = e<8 ? 8*hi+e : 16+8*hi+(e-8)  -> two contiguous
// 8-halfword (16 B) runs per 32-wide K step for both A and B.
// --------------------------------------------------------------------------
__device__ __forceinline__ v16h load_frag(const _Float16* __restrict__ base,
                                          int k0, int hi) {
    v8h lo = *(const v8h*)(base + k0 + 8 * hi);
    v8h hh = *(const v8h*)(base + k0 + 16 + 8 * hi);
    v16h f;
#pragma unroll
    for (int e = 0; e < 8; ++e) { f[e] = lo[e]; f[8 + e] = hh[e]; }
    return f;
}

template <int K>
__global__ void gemm_wmma_f16(const _Float16* __restrict__ Xh,
                              const _Float16* __restrict__ Wt,
                              float* __restrict__ Y, int M, int N,
                              const float* __restrict__ bias) {
    int wave   = (blockIdx.x * blockDim.x + threadIdx.x) >> 5;
    int lane   = threadIdx.x & 31;
    int tilesN = N >> 4;
    int nTiles = (M >> 4) * tilesN;
    if (wave >= nTiles) return;
    int tm = wave / tilesN;
    int tn = wave % tilesN;
    int m  = lane & 15;
    int hi = lane >> 4;

    const _Float16* __restrict__ arow = Xh + (size_t)(tm * 16 + m) * K;
    int ncol = tn * 16 + m;                               // B/C/D column
    const _Float16* __restrict__ bcol = Wt + (size_t)ncol * K;

    v8f acc = {};
#pragma unroll
    for (int k0 = 0; k0 < K; k0 += 32) {
        v16h a = load_frag(arow, k0, hi);
        v16h b = load_frag(bcol, k0, hi);
        acc = __builtin_amdgcn_wmma_f32_16x16x32_f16(
            false, a, false, b, (short)0, acc, false, false);
    }

    float bc = bias ? bias[ncol] : 0.0f;
#pragma unroll
    for (int r = 0; r < 8; ++r) {
        int row = tm * 16 + r + 8 * hi;                   // C/D: VGPR r -> M=r+8*hi
        Y[(size_t)row * N + ncol] = acc[r] + bc;
    }
}

// --------------------------------------------------------------------------
// Zero an AGG buffer (float4 vectorized, grid-stride)
// --------------------------------------------------------------------------
__global__ void zero_buf(float* __restrict__ p, int n) {
    int stride = gridDim.x * blockDim.x;
    v4f z = {0.f, 0.f, 0.f, 0.f};
    for (int i = blockIdx.x * blockDim.x + threadIdx.x; i < (n >> 2); i += stride)
        ((v4f*)p)[i] = z;
}

// --------------------------------------------------------------------------
// Edge aggregation: AGG[dst] += H[src] * (dinv[src]*w*dinv[dst]).
// One wave per edge; 32 lanes sweep the 256 features (coalesced loads,
// f32 atomic adds; AGG is 8 MB << 192 MB L2 so atomics resolve in L2).
// Edges [0,E) are real, [E, E+N) are self-loops with w=1.
// --------------------------------------------------------------------------
__global__ void aggregate(const float* __restrict__ Hm, const int* __restrict__ src,
                          const int* __restrict__ dst, const float* __restrict__ w,
                          const float* __restrict__ dinv, float* __restrict__ AGG) {
    int eidx = (blockIdx.x * blockDim.x + threadIdx.x) >> 5;
    int lane = threadIdx.x & 31;
    const int ET = NEDGE + NODES;
    if (eidx >= ET) return;
    int s, d; float wt;
    if (eidx < NEDGE) { s = src[eidx]; d = dst[eidx]; wt = w[eidx]; }
    else              { s = d = eidx - NEDGE; wt = 1.0f; }
    float norm = dinv[s] * wt * dinv[d];
    const float* __restrict__ hs = Hm + (size_t)s * HDIM;
    float* __restrict__ ad = AGG + (size_t)d * HDIM;
#pragma unroll
    for (int f = lane; f < HDIM; f += 32)
        atomicAdd(ad + f, hs[f] * norm);
}

// --------------------------------------------------------------------------
// Xh = (f16) relu(AGG + b)  -- writes the next GEMM's A operand directly
// --------------------------------------------------------------------------
__global__ void relu_bias_f16(const float* __restrict__ AGG, const float* __restrict__ b,
                              _Float16* __restrict__ Xh) {
    int i = blockIdx.x * blockDim.x + threadIdx.x;
    if (i < NODES * HDIM) {
        float v = AGG[i] + b[i & (HDIM - 1)];
        Xh[i] = (_Float16)(v > 0.0f ? v : 0.0f);
    }
}

// --------------------------------------------------------------------------
extern "C" void kernel_launch(void* const* d_in, const int* in_sizes, int n_in,
                              void* d_out, int out_size, void* d_ws, size_t ws_size,
                              hipStream_t stream) {
    const float* z    = (const float*)d_in[0];
    const int*   eidx = (const int*)  d_in[1];
    const float* ea   = (const float*)d_in[2];
    const float* W0   = (const float*)d_in[3];
    const float* b0   = (const float*)d_in[4];
    const float* W1   = (const float*)d_in[5];
    const float* b1   = (const float*)d_in[6];
    const float* linW = (const float*)d_in[7];
    const float* linb = (const float*)d_in[8];
    const int* srcI = eidx;
    const int* dstI = eidx + NEDGE;

    float* out    = (float*)d_out;                 // [NODES, XFEAT] fp32
    int*   new_ei = (int*)d_out + NODES * XFEAT;   // [2, N*N] int32 (iota)

    // workspace layout
    char* ws = (char*)d_ws;
    float*     dinv = (float*)ws;                                   //  32 KB
    _Float16*  Wt0  = (_Float16*)(ws + (32 << 10));                 //  64 KB  [256,128]
    _Float16*  Wt1  = (_Float16*)(ws + (96 << 10));                 // 128 KB  [256,256]
    _Float16*  WtL  = (_Float16*)(ws + (224 << 10));                //  64 KB  [128,256]
    _Float16*  Xh   = (_Float16*)(ws + (288 << 10));                //   4 MB  [8192,256]
    float*     bufH = (float*)(ws + (288 << 10) + (4u << 20));      //   8 MB
    float*     bufAgg = bufH + (size_t)NODES * HDIM;                //   8 MB

    // Static second output (the 8192x8192 sigmoid matmul feeding it is dead code)
    fill_edge_index<<<4096, 256, 0, stream>>>(new_ei);

    // deg -> dinv
    deg_init    <<<NODES / 256, 256, 0, stream>>>(dinv);
    deg_accum   <<<NEDGE / 256, 256, 0, stream>>>(dstI, ea, dinv);
    deg_finalize<<<NODES / 256, 256, 0, stream>>>(dinv);

    // Weight pre-conversion to f16 col-major (one-shot, tiny)
    cvt_wt_f16T<<<(ZFEAT * HDIM) / 256, 256, 0, stream>>>(W0,   Wt0, ZFEAT, HDIM);
    cvt_wt_f16T<<<(HDIM * HDIM)  / 256, 256, 0, stream>>>(W1,   Wt1, HDIM,  HDIM);
    cvt_wt_f16T<<<(HDIM * XFEAT) / 256, 256, 0, stream>>>(linW, WtL, HDIM,  XFEAT);

    // z -> f16 A operand
    cvt_f16<<<(NODES * ZFEAT) / 256, 256, 0, stream>>>(z, Xh, NODES * ZFEAT);

    auto blocks_for = [](int M, int N) { return ((M / 16) * (N / 16) + 3) / 4; };

    // Layer 0: H = z @ W0 ; AGG ; relu(+b0) -> f16
    gemm_wmma_f16<ZFEAT><<<blocks_for(NODES, HDIM), 128, 0, stream>>>(
        Xh, Wt0, bufH, NODES, HDIM, nullptr);
    zero_buf <<<2048, 256, 0, stream>>>(bufAgg, NODES * HDIM);
    aggregate<<<(NEDGE + NODES) / 8, 256, 0, stream>>>(bufH, srcI, dstI, ea, dinv, bufAgg);
    relu_bias_f16<<<NODES * HDIM / 256, 256, 0, stream>>>(bufAgg, b0, Xh);

    // Layer 1: H = X1 @ W1 ; AGG ; relu(+b1) -> f16
    gemm_wmma_f16<HDIM><<<blocks_for(NODES, HDIM), 128, 0, stream>>>(
        Xh, Wt1, bufH, NODES, HDIM, nullptr);
    zero_buf <<<2048, 256, 0, stream>>>(bufAgg, NODES * HDIM);
    aggregate<<<(NEDGE + NODES) / 8, 256, 0, stream>>>(bufH, srcI, dstI, ea, dinv, bufAgg);
    relu_bias_f16<<<NODES * HDIM / 256, 256, 0, stream>>>(bufAgg, b1, Xh);

    // Final projection with bias, straight into d_out
    gemm_wmma_f16<HDIM><<<blocks_for(NODES, XFEAT), 128, 0, stream>>>(
        Xh, WtL, out, NODES, XFEAT, linb);
}
